// PointcloudEncoder_37598143709547
// MI455X (gfx1250) — matrix operand order repaired and checked by reading, not compiled, not run on hardware
//
#include <hip/hip_runtime.h>

#define BB 8
#define NN 8192
#define GG 512
#define MM 32
#define ENC 512
#define TD 768

typedef __attribute__((ext_vector_type(16))) __bf16 v16bf;
typedef __attribute__((ext_vector_type(8)))  float  v8f;
typedef __attribute__((ext_vector_type(4)))  unsigned int v4u;

union Frag {
    v16bf  v;
    __bf16 h[16];
    v4u    q[2];
};

// ---- fragment loaders (CDNA5 wave32 WMMA layouts) ----

// B fragment (32x16 KxN, bf16): lane n=lane&15 holds column n; elements e<8 -> K = kbase+koff+e,
// e>=8 -> K = kbase+16+koff+(e-8), koff = (lane&16)?8:0 (same order as A).
// Since B = W^T (out[m,o] = sum_c x[m,c]*w[o,c]), each lane reads two 16B chunks from weight row n.
__device__ inline v16bf ld_bfrag(const __bf16* wrow, int kbase, int kh) {
    Frag f;
    const __bf16* p = wrow + kbase + (kh ? 8 : 0);
    f.q[0] = *(const v4u*)(p);
    f.q[1] = *(const v4u*)(p + 16);
    return f.v;
}

// A fragment (16x32 MxK, bf16): lane row m=lane&15; elements e<8 -> K = kbase+koff+e,
// e>=8 -> K = kbase+16+koff+(e-8), koff = (lane&16)?8:0. Two 16B chunks.
__device__ inline v16bf ld_afrag(const __bf16* rowp, int kbase, int kh) {
    int k0 = kbase + (kh ? 8 : 0);
    Frag f;
    f.q[0] = *(const v4u*)(rowp + k0);
    f.q[1] = *(const v4u*)(rowp + k0 + 16);
    return f.v;
}

// A fragment synthesized from a broadcast f32 vector (same value for every row m)
__device__ inline v16bf ld_afrag_bcast(const float* src, int kbase, int kh) {
    int k0 = kbase + (kh ? 8 : 0);
    Frag f;
#pragma unroll
    for (int e = 0; e < 8; ++e) {
        f.h[e]     = (__bf16)src[k0 + e];
        f.h[e + 8] = (__bf16)src[k0 + 16 + e];
    }
    return f.v;
}

__device__ inline v8f wmma_bf16(v16bf a, v16bf b, v8f c) {
    return __builtin_amdgcn_wmma_f32_16x16x32_bf16(false, a, false, b, (short)0, c, false, false);
}

// ============================ prep kernels ============================

__global__ void prep_small_kernel(const float* w1, const float* b1, const float* g1,
                                  const float* be1, const float* m1, const float* v1,
                                  const float* b3, const float* g3, const float* be3,
                                  const float* m3, const float* v3,
                                  float* w1f, float* b1f, float* a3f, float* b3f) {
    int t = threadIdx.x;
    for (int o = t; o < 128; o += 256) {
        float a = g1[o] * rsqrtf(v1[o] + 1e-5f);
        b1f[o] = b1[o] * a + be1[o] - m1[o] * a;
#pragma unroll
        for (int c = 0; c < 6; ++c) w1f[o * 6 + c] = w1[o * 6 + c] * a;
    }
    for (int o = t; o < 512; o += 256) {
        float a = g3[o] * rsqrtf(v3[o] + 1e-5f);
        a3f[o] = a;
        b3f[o] = b3[o] * a + be3[o] - m3[o] * a;
    }
}

__global__ void cvt_bf16_kernel(const float* src, __bf16* dst, const float* rowscale,
                                int rows, int cols) {
    int i = blockIdx.x * blockDim.x + threadIdx.x;
    if (i < rows * cols) {
        float s = rowscale ? rowscale[i / cols] : 1.0f;
        dst[i] = (__bf16)(src[i] * s);
    }
}

// ============================ FPS ============================
// 1 block per batch, 1024 threads; points + running min-dist in registers.
__global__ __launch_bounds__(1024) void fps_kernel(const float* pts, float* centers) {
    __shared__ float redv[1024];
    __shared__ int   redi[1024];
    __shared__ float s_p[3];
    __shared__ int   s_sel;
    int b = blockIdx.x, t = threadIdx.x;
    const float* P = pts + (size_t)b * NN * 3;

    float px[8], py[8], pz[8], md[8];
#pragma unroll
    for (int r = 0; r < 8; ++r) {
        int i = t * 8 + r;
        px[r] = P[i * 3 + 0];
        py[r] = P[i * 3 + 1];
        pz[r] = P[i * 3 + 2];
        md[r] = 1e10f;
    }
    if (t == 0) {
        s_sel = 0;
        s_p[0] = P[0]; s_p[1] = P[1]; s_p[2] = P[2];
        centers[((size_t)b * GG + 0) * 3 + 0] = P[0];
        centers[((size_t)b * GG + 0) * 3 + 1] = P[1];
        centers[((size_t)b * GG + 0) * 3 + 2] = P[2];
    }
    __syncthreads();

    for (int j = 1; j < GG; ++j) {
        float qx = s_p[0], qy = s_p[1], qz = s_p[2];
        float bestv = -1.0f;
        int   besti = t * 8;
#pragma unroll
        for (int r = 0; r < 8; ++r) {
            float dx = px[r] - qx, dy = py[r] - qy, dz = pz[r] - qz;
            float d = dx * dx + dy * dy + dz * dz;
            float mv = fminf(md[r], d);
            md[r] = mv;
            if (mv > bestv) { bestv = mv; besti = t * 8 + r; }
        }
        redv[t] = bestv; redi[t] = besti;
        __syncthreads();
        for (int s = 512; s > 0; s >>= 1) {
            if (t < s) {
                float v2 = redv[t + s]; int i2 = redi[t + s];
                if (v2 > redv[t] || (v2 == redv[t] && i2 < redi[t])) { redv[t] = v2; redi[t] = i2; }
            }
            __syncthreads();
        }
        if (t == 0) {
            int ni = redi[0];
            s_sel = ni;
            s_p[0] = P[ni * 3 + 0]; s_p[1] = P[ni * 3 + 1]; s_p[2] = P[ni * 3 + 2];
            centers[((size_t)b * GG + j) * 3 + 0] = s_p[0];
            centers[((size_t)b * GG + j) * 3 + 1] = s_p[1];
            centers[((size_t)b * GG + j) * 3 + 2] = s_p[2];
        }
        __syncthreads();
    }
    (void)s_sel;
}

// ============================ KNN ============================
// 1 block per (b,g); 32 extract-min rounds (tie-break lowest index, top_k stable).
__global__ __launch_bounds__(256) void knn_kernel(const float* pts, const float* centers, int* knn) {
    __shared__ float dist[NN];
    __shared__ float redv[256];
    __shared__ int   redi[256];
    int blk = blockIdx.x;
    int b = blk >> 9, g = blk & (GG - 1);
    int t = threadIdx.x;
    const float* P = pts + (size_t)b * NN * 3;
    float cx = centers[((size_t)b * GG + g) * 3 + 0];
    float cy = centers[((size_t)b * GG + g) * 3 + 1];
    float cz = centers[((size_t)b * GG + g) * 3 + 2];
    for (int i = t; i < NN; i += 256) {
        float dx = P[i * 3] - cx, dy = P[i * 3 + 1] - cy, dz = P[i * 3 + 2] - cz;
        dist[i] = dx * dx + dy * dy + dz * dz;
    }
    __syncthreads();
    int base = t * 32;
    for (int r = 0; r < MM; ++r) {
        float bv = 3.4e38f; int bi = base;
#pragma unroll 4
        for (int q = 0; q < 32; ++q) {
            float v = dist[base + q];
            if (v < bv) { bv = v; bi = base + q; }
        }
        redv[t] = bv; redi[t] = bi;
        __syncthreads();
        for (int s = 128; s > 0; s >>= 1) {
            if (t < s) {
                float v2 = redv[t + s]; int i2 = redi[t + s];
                if (v2 < redv[t] || (v2 == redv[t] && i2 < redi[t])) { redv[t] = v2; redi[t] = i2; }
            }
            __syncthreads();
        }
        if (t == 0) { knn[(size_t)blk * MM + r] = redi[0]; dist[redi[0]] = 3.4e38f; }
        __syncthreads();
    }
}

// ============================ Encoder mega-kernel ============================
// 1 block per (b,g) group; 8 waves. All activations in LDS; lin2/3/4 on WMMA bf16.
__global__ __launch_bounds__(256) void encoder_kernel(
    const float* pts, const float* colors, const float* centers, const int* knn,
    const float* w1f, const float* b1f,
    const __bf16* w2bf, const float* b2,
    const __bf16* w3bf, const float* b3f,
    const __bf16* w4bf, const float* b4,
    float* tokens) {
    __shared__ __bf16 s_h1[32 * 128];   // relu(bn(lin1))
    __shared__ __bf16 s_y2[32 * 256];   // lin2 output (concat lower half)
    __shared__ __bf16 s_h3[32 * 512];   // relu(bn(lin3))
    __shared__ float  s_hg[256];        // column max of lin2 (concat broadcast half)
    __shared__ float  s_feats[32 * 6];
    __shared__ int    s_idx[32];

    int blk = blockIdx.x;
    int b = blk >> 9, g = blk & (GG - 1);
    int t = threadIdx.x, lane = t & 31, wave = t >> 5;
    int ln = lane & 15, kh = lane & 16;

    if (t < 32) s_idx[t] = knn[(size_t)blk * MM + t];
    __syncthreads();
    if (t < 192) {
        int m = t / 6, c = t % 6;
        int pi = s_idx[m];
        float v;
        if (c < 3) v = pts[((size_t)b * NN + pi) * 3 + c] - centers[((size_t)b * GG + g) * 3 + c];
        else       v = colors[((size_t)b * NN + pi) * 3 + (c - 3)];
        s_feats[m * 6 + c] = v;
    }
    __syncthreads();

    // ---- lin1 (K=6) + folded BN + ReLU, plain VALU ----
#pragma unroll
    for (int r = 0; r < 16; ++r) {
        int e = t * 16 + r;         // 0..4095
        int m = e >> 7, o = e & 127;
        float acc = b1f[o];
#pragma unroll
        for (int c = 0; c < 6; ++c) acc += s_feats[m * 6 + c] * w1f[o * 6 + c];
        s_h1[m * 128 + o] = (__bf16)fmaxf(acc, 0.0f);
    }
    __syncthreads();

    // ---- lin2 (128->256) WMMA; also compute per-column max (hg) ----
    for (int nn2 = 0; nn2 < 2; ++nn2) {
        int nt = wave * 2 + nn2;
        int n = nt * 16 + ln;
        v8f a0 = {}, a1 = {};
#pragma unroll
        for (int ks = 0; ks < 4; ++ks) {
            v16bf bf = ld_bfrag(w2bf + n * 128, ks * 32, kh);
            a0 = wmma_bf16(ld_afrag(s_h1 + ln * 128,        ks * 32, kh), bf, a0);
            a1 = wmma_bf16(ld_afrag(s_h1 + (16 + ln) * 128, ks * 32, kh), bf, a1);
        }
        float bias = b2[n];
        float cm = -3.4e38f;
        int rb = kh ? 8 : 0;
#pragma unroll
        for (int i = 0; i < 8; ++i) {
            float v0 = a0[i] + bias, v1 = a1[i] + bias;
            s_y2[(rb + i) * 256 + n]      = (__bf16)v0;
            s_y2[(16 + rb + i) * 256 + n] = (__bf16)v1;
            cm = fmaxf(cm, fmaxf(v0, v1));
        }
        cm = fmaxf(cm, __shfl_xor(cm, 16, 32));
        if (!kh) s_hg[n] = cm;
    }
    __syncthreads();

    // ---- lin3 (concat 512 -> 512), folded BN + ReLU; broadcast half from s_hg ----
    for (int q = 0; q < 4; ++q) {
        int nt = wave * 4 + q, n = nt * 16 + ln;
        v8f a0 = {}, a1 = {};
#pragma unroll
        for (int ks = 0; ks < 8; ++ks) {       // K 0..255: broadcast hg (same for all rows)
            v16bf bf = ld_bfrag(w3bf + n * 512, ks * 32, kh);
            v16bf xh = ld_afrag_bcast(s_hg, ks * 32, kh);
            a0 = wmma_bf16(xh, bf, a0);
            a1 = wmma_bf16(xh, bf, a1);
        }
#pragma unroll
        for (int ks = 0; ks < 8; ++ks) {       // K 256..511: per-point lin2 output
            v16bf bf = ld_bfrag(w3bf + n * 512 + 256, ks * 32, kh);
            a0 = wmma_bf16(ld_afrag(s_y2 + ln * 256,        ks * 32, kh), bf, a0);
            a1 = wmma_bf16(ld_afrag(s_y2 + (16 + ln) * 256, ks * 32, kh), bf, a1);
        }
        float bias = b3f[n];
        int rb = kh ? 8 : 0;
#pragma unroll
        for (int i = 0; i < 8; ++i) {
            s_h3[(rb + i) * 512 + n]      = (__bf16)fmaxf(a0[i] + bias, 0.0f);
            s_h3[(16 + rb + i) * 512 + n] = (__bf16)fmaxf(a1[i] + bias, 0.0f);
        }
    }
    __syncthreads();

    // ---- lin4 (512 -> 512) + maxpool over the 32 points ----
    for (int q = 0; q < 4; ++q) {
        int nt = wave * 4 + q, n = nt * 16 + ln;
        v8f a0 = {}, a1 = {};
#pragma unroll
        for (int ks = 0; ks < 16; ++ks) {
            v16bf bf = ld_bfrag(w4bf + n * 512, ks * 32, kh);
            a0 = wmma_bf16(ld_afrag(s_h3 + ln * 512,        ks * 32, kh), bf, a0);
            a1 = wmma_bf16(ld_afrag(s_h3 + (16 + ln) * 512, ks * 32, kh), bf, a1);
        }
        float cm = -3.4e38f;
#pragma unroll
        for (int i = 0; i < 8; ++i) cm = fmaxf(cm, fmaxf(a0[i], a1[i]));
        cm += b4[n];                                  // max(x+b) == max(x)+b (b per column)
        cm = fmaxf(cm, __shfl_xor(cm, 16, 32));       // combine row halves
        if (!kh) tokens[(size_t)blk * 512 + n] = cm;
    }
}

// ============================ Head ============================
// tokens @ wt^T + bt + pos (gelu GEMM fused as extra WMMA K-steps) -> out[:,1:,:]
__global__ __launch_bounds__(256) void head_kernel(
    const float* tokens, const float* centers,
    const __bf16* wtbf, const float* bt,
    const float* wp1, const float* bp1,
    const __bf16* wp2bf, const float* bp2,
    float* out) {
    __shared__ __bf16 s_A[32 * 512];   // 32 token rows, bf16
    __shared__ __bf16 s_t[32 * 128];   // gelu(lin(center)) features, bf16

    int b = blockIdx.x >> 4, rt = blockIdx.x & 15;
    int g0 = rt * 32;
    int t = threadIdx.x, lane = t & 31, wave = t >> 5;
    int ln = lane & 15, kh = lane & 16;

#pragma unroll
    for (int r = 0; r < 64; ++r) {
        int e = t * 64 + r;            // 0..16383
        int m = e >> 9, c = e & 511;
        s_A[e] = (__bf16)tokens[((size_t)b * GG + g0 + m) * 512 + c];
    }
#pragma unroll
    for (int r = 0; r < 16; ++r) {
        int e = t * 16 + r;            // 0..4095
        int m = e >> 7, o = e & 127;
        const float* C = centers + ((size_t)b * GG + g0 + m) * 3;
        float x = bp1[o] + wp1[o * 3] * C[0] + wp1[o * 3 + 1] * C[1] + wp1[o * 3 + 2] * C[2];
        float ge = 0.5f * x * (1.0f + erff(x * 0.70710678118f));
        s_t[m * 128 + o] = (__bf16)ge;
    }
    __syncthreads();

    for (int q = 0; q < 6; ++q) {
        int nt = wave * 6 + q, n = nt * 16 + ln;
        v8f a0 = {}, a1 = {};
#pragma unroll
        for (int ks = 0; ks < 16; ++ks) {            // tokens @ wt^T, K=512
            v16bf bf = ld_bfrag(wtbf + n * 512, ks * 32, kh);
            a0 = wmma_bf16(ld_afrag(s_A + ln * 512,        ks * 32, kh), bf, a0);
            a1 = wmma_bf16(ld_afrag(s_A + (16 + ln) * 512, ks * 32, kh), bf, a1);
        }
#pragma unroll
        for (int ks = 0; ks < 4; ++ks) {             // + gelu_feats @ wp2^T, K=128
            v16bf bf = ld_bfrag(wp2bf + n * 128, ks * 32, kh);
            a0 = wmma_bf16(ld_afrag(s_t + ln * 128,        ks * 32, kh), bf, a0);
            a1 = wmma_bf16(ld_afrag(s_t + (16 + ln) * 128, ks * 32, kh), bf, a1);
        }
        float bias = bt[n] + bp2[n];
        int rb = kh ? 8 : 0;
#pragma unroll
        for (int i = 0; i < 8; ++i) {
            int r0 = rb + i, r1 = 16 + rb + i;
            out[((size_t)b * (GG + 1) + 1 + g0 + r0) * TD + n] = a0[i] + bias;
            out[((size_t)b * (GG + 1) + 1 + g0 + r1) * TD + n] = a1[i] + bias;
        }
    }
}

__global__ void cls_kernel(const float* cls_tok, const float* cls_pos, float* out) {
    int i = blockIdx.x * blockDim.x + threadIdx.x;
    if (i < BB * TD) {
        int b = i / TD, o = i % TD;
        out[((size_t)b * (GG + 1)) * TD + o] = cls_tok[o] + cls_pos[o];
    }
}

// ============================ launch ============================

extern "C" void kernel_launch(void* const* d_in, const int* in_sizes, int n_in,
                              void* d_out, int out_size, void* d_ws, size_t ws_size,
                              hipStream_t stream) {
    const float* pts    = (const float*)d_in[0];
    const float* colors = (const float*)d_in[1];
    const float* w1  = (const float*)d_in[2];  const float* b1  = (const float*)d_in[3];
    const float* g1  = (const float*)d_in[4];  const float* be1 = (const float*)d_in[5];
    const float* m1  = (const float*)d_in[6];  const float* v1  = (const float*)d_in[7];
    const float* w2  = (const float*)d_in[8];  const float* b2  = (const float*)d_in[9];
    const float* w3  = (const float*)d_in[10]; const float* b3  = (const float*)d_in[11];
    const float* g3  = (const float*)d_in[12]; const float* be3 = (const float*)d_in[13];
    const float* m3  = (const float*)d_in[14]; const float* v3  = (const float*)d_in[15];
    const float* w4  = (const float*)d_in[16]; const float* b4  = (const float*)d_in[17];
    const float* wt  = (const float*)d_in[18]; const float* bt  = (const float*)d_in[19];
    const float* wp1 = (const float*)d_in[20]; const float* bp1 = (const float*)d_in[21];
    const float* wp2 = (const float*)d_in[22]; const float* bp2 = (const float*)d_in[23];
    const float* cls_tok = (const float*)d_in[24];
    const float* cls_pos = (const float*)d_in[25];
    float* out = (float*)d_out;

    char* ws = (char*)d_ws;
    size_t off = 0;
    auto alloc = [&](size_t bytes) -> void* {
        void* p = ws + off;
        off = (off + bytes + 255) & ~(size_t)255;
        return p;
    };
    float*  centers = (float*)alloc((size_t)BB * GG * 3 * 4);
    int*    knn     = (int*)  alloc((size_t)BB * GG * MM * 4);
    float*  tokens  = (float*)alloc((size_t)BB * GG * 512 * 4);
    float*  w1f     = (float*)alloc(128 * 6 * 4);
    float*  b1f     = (float*)alloc(128 * 4);
    float*  a3f     = (float*)alloc(512 * 4);
    float*  b3f     = (float*)alloc(512 * 4);
    __bf16* w2bf    = (__bf16*)alloc(256 * 128 * 2);
    __bf16* w3bf    = (__bf16*)alloc(512 * 512 * 2);
    __bf16* w4bf    = (__bf16*)alloc(512 * 512 * 2);
    __bf16* wtbf    = (__bf16*)alloc((size_t)TD * 512 * 2);
    __bf16* wp2bf   = (__bf16*)alloc((size_t)TD * 128 * 2);
    (void)in_sizes; (void)n_in; (void)out_size; (void)ws_size;

    prep_small_kernel<<<1, 256, 0, stream>>>(w1, b1, g1, be1, m1, v1,
                                             b3, g3, be3, m3, v3,
                                             w1f, b1f, a3f, b3f);
    cvt_bf16_kernel<<<(256 * 128 + 255) / 256, 256, 0, stream>>>(w2, w2bf, nullptr, 256, 128);
    cvt_bf16_kernel<<<(512 * 512 + 255) / 256, 256, 0, stream>>>(w3, w3bf, a3f, 512, 512);
    cvt_bf16_kernel<<<(512 * 512 + 255) / 256, 256, 0, stream>>>(w4, w4bf, nullptr, 512, 512);
    cvt_bf16_kernel<<<(TD * 512 + 255) / 256, 256, 0, stream>>>(wt, wtbf, nullptr, TD, 512);
    cvt_bf16_kernel<<<(TD * 128 + 255) / 256, 256, 0, stream>>>(wp2, wp2bf, nullptr, TD, 128);

    fps_kernel<<<BB, 1024, 0, stream>>>(pts, centers);
    knn_kernel<<<BB * GG, 256, 0, stream>>>(pts, centers, knn);
    encoder_kernel<<<BB * GG, 256, 0, stream>>>(pts, colors, centers, knn,
                                                w1f, b1f, w2bf, b2, w3bf, b3f, w4bf, b4, tokens);
    head_kernel<<<BB * (GG / 32), 256, 0, stream>>>(tokens, centers, wtbf, bt,
                                                    wp1, bp1, wp2bf, bp2, out);
    cls_kernel<<<(BB * TD + 255) / 256, 256, 0, stream>>>(cls_tok, cls_pos, out);
}
